// MPNNBlock_65335042506810
// MI455X (gfx1250) — matrix operand order repaired and checked by reading, not compiled, not run on hardware
//
#include <hip/hip_runtime.h>
#include <hip/hip_bf16.h>
#include <math.h>

typedef __bf16 bf16_t;
typedef __bf16 v16bf __attribute__((ext_vector_type(16)));
typedef float  v8f   __attribute__((ext_vector_type(8)));

#define NNODES 8192
#define KNB 32
#define ENB 16
#define ROWS 48      // feat rows per node
#define GNODE 2      // nodes per workgroup
#define ROWS2 96     // GNODE * ROWS
#define DD 256
#define HH 512
#define CINW 768

#define FS 784   // feats LDS stride (bf16 elems), 1568B, 16B aligned
#define AS 528   // act   LDS stride (bf16 elems), 1056B, 16B aligned
#define US 260   // upd   LDS stride (f32 elems)
#define XS 272   // x     LDS stride (bf16 elems), 544B, 16B aligned

// pass-kernel LDS map (dynamic):
//   feats : [0, 150528)            96 x FS bf16
//   act   : [150528, 251904)       96 x AS bf16
//   upd   : overlays feats         96 x US f32 (99840 B)
//   rmask : [251904, 252288)       96 f32
//   red   : [252288, 252352)       16 f32
#define SMEM_PASS 252352
#define SMEM_MLP  84480

__device__ __forceinline__ float gelu_tanh(float x) {
    float x3 = x * x * x;
    return 0.5f * x * (1.0f + tanhf(0.7978845608028654f * (x + 0.044715f * x3)));
}

__device__ __forceinline__ float wave_sum(float v) {
    v += __shfl_xor(v, 16, 32);
    v += __shfl_xor(v, 8, 32);
    v += __shfl_xor(v, 4, 32);
    v += __shfl_xor(v, 2, 32);
    v += __shfl_xor(v, 1, 32);
    return v;
}

union FragU { uint4 u[2]; v16bf v; };

// A-matrix 16x32 bf16 fragment from LDS (row-major, stride elems).
__device__ __forceinline__ v16bf load_a_frag(const bf16_t* base, int stride,
                                             int mbase, int kbase, int l, int hi) {
    const bf16_t* p = base + (mbase + l) * stride + kbase + hi * 8;
    FragU f;
    f.u[0] = *(const uint4*)p;
    f.u[1] = *(const uint4*)(p + 16);
    return f.v;
}

// B-matrix 32x16 bf16 fragment from global weight stored [N][Ktot] (K contiguous).
__device__ __forceinline__ v16bf load_b_frag(const bf16_t* __restrict__ W, int ktot,
                                             int nbase, int kbase, int l, int hi) {
    const bf16_t* p = W + (size_t)(nbase + l) * ktot + kbase + hi * 8;
    FragU f;
    f.u[0] = *(const uint4*)p;
    f.u[1] = *(const uint4*)(p + 16);
    return f.v;
}

__device__ __forceinline__ v8f wmma_bf16(v16bf a, v16bf b, v8f c) {
    return __builtin_amdgcn_wmma_f32_16x16x32_bf16(false, a, false, b,
                                                   (short)0, c, false, false);
}

// ---------------------------------------------------------------------------
// Weight convert + transpose: src [in_dim][out_dim] f32 -> dst [out_dim][in_dim] bf16
// ---------------------------------------------------------------------------
__global__ void convert_transpose_kernel(const float* __restrict__ src,
                                         bf16_t* __restrict__ dst,
                                         int in_dim, int out_dim) {
    int idx = blockIdx.x * blockDim.x + threadIdx.x;
    if (idx >= in_dim * out_dim) return;
    int o = idx / in_dim, ii = idx - o * in_dim;
    dst[idx] = (bf16_t)src[(size_t)ii * out_dim + o];
}

// ---------------------------------------------------------------------------
// Two nodes per workgroup; 256 threads = 8 waves.
// Each wave owns N-column strips and sweeps all 6 M-tiles inside the k-loop,
// so every weight element is read from L2 exactly once per workgroup.
// phase 0: pair/extra update (gated MLP 1) -> LN -> d_out pair/extra regions
// phase 1: message (gated MLP 2) -> masked mean -> LN -> local1
// ---------------------------------------------------------------------------
__global__ __launch_bounds__(256) void mpnn_pass_kernel(
    const float* __restrict__ local_in,      // [N,256]
    const float* __restrict__ pair_in,       // [N,32,256]
    const float* __restrict__ extra_in,      // [N,16,256]
    const int*   __restrict__ neighbours,    // [N,32]
    const unsigned char* __restrict__ mask,        // [N]
    const unsigned char* __restrict__ extra_mask,  // [N,16]
    const bf16_t* __restrict__ Wept,   // [256][256]
    const bf16_t* __restrict__ Wgt,    // [512][768]
    const bf16_t* __restrict__ Wht,    // [512][768]
    const bf16_t* __restrict__ Wot,    // [256][512]
    const float* __restrict__ bg, const float* __restrict__ bh,
    const float* __restrict__ bo,
    const float* __restrict__ ln_s, const float* __restrict__ ln_b,
    float* __restrict__ out_pair, float* __restrict__ out_extra,
    float* __restrict__ out_local1,
    int phase)
{
    extern __shared__ __align__(16) char smem[];
    bf16_t* feats = (bf16_t*)smem;
    bf16_t* act   = (bf16_t*)(smem + 150528);
    float*  upd   = (float*)smem;                 // overlays feats after GEMM1
    float*  rmask = (float*)(smem + 251904);
    float*  red   = (float*)(smem + 252288);

    const int i0   = blockIdx.x * GNODE;
    const int tid  = threadIdx.x;
    const int wave = tid >> 5;
    const int lane = tid & 31;
    const int l    = lane & 15;
    const int hi   = lane >> 4;

    // ---- stage feats (bf16) ----
    // cols [0,256): full_pair for both nodes
    for (int e = tid; e < ROWS2 * DD; e += 256) {
        int r = e >> 8, c = e & 255;
        int g = (r >= ROWS) ? 1 : 0;
        int rr = r - g * ROWS;
        int i = i0 + g;
        float v = (rr < KNB) ? pair_in[((size_t)i * KNB + rr) * DD + c]
                             : extra_in[((size_t)i * ENB + (rr - KNB)) * DD + c];
        feats[r * FS + c] = (bf16_t)v;
    }
    // cols [256,512): center broadcast
    {
        bf16_t b0 = (bf16_t)local_in[(size_t)i0 * DD + tid];
        bf16_t b1 = (bf16_t)local_in[(size_t)(i0 + 1) * DD + tid];
        for (int r = 0; r < ROWS; ++r) {
            feats[r * FS + DD + tid] = b0;
            feats[(ROWS + r) * FS + DD + tid] = b1;
        }
    }
    // cols [512,768) rows [g*48, g*48+32): gathered neighbours
    for (int e = tid; e < GNODE * KNB * DD; e += 256) {
        int r = e >> 8, c = e & 255;          // r in 0..63
        int g = r >> 5, rr = r & 31;
        int nbv = neighbours[(size_t)(i0 + g) * KNB + rr];
        if (nbv < 0) nbv += NNODES;
        feats[(g * ROWS + rr) * FS + 2 * DD + c] = (bf16_t)local_in[(size_t)nbv * DD + c];
    }
    __syncthreads();

    // ---- extra rows of full_nb: feats[g*48+32 : g*48+48, 0:256] @ Wept
    //      -> feats[..., 512:768].  B shared across both nodes.
    for (int ntI = 0; ntI < 2; ++ntI) {
        int ncol = (wave + 8 * ntI) * 16;     // 16 N-tiles over 8 waves x 2
        v8f acc0 = {}, acc1 = {};
        for (int kb = 0; kb < DD; kb += 32) {
            v16bf b  = load_b_frag(Wept, DD, ncol, kb, l, hi);
            v16bf a0 = load_a_frag(feats, FS, KNB, kb, l, hi);
            acc0 = wmma_bf16(a0, b, acc0);
            v16bf a1 = load_a_frag(feats, FS, ROWS + KNB, kb, l, hi);
            acc1 = wmma_bf16(a1, b, acc1);
        }
        #pragma unroll
        for (int p = 0; p < 8; ++p) {
            feats[(KNB + p + hi * 8) * FS + 2 * DD + ncol + l] = (bf16_t)acc0[p];
            feats[(ROWS + KNB + p + hi * 8) * FS + 2 * DD + ncol + l] = (bf16_t)acc1[p];
        }
    }
    __syncthreads();

    // ---- GEMM 1: act = gelu(feats@Wg + bg) * (feats@Wh + bh)
    //      wave owns N-strip; 6 M-tiles reuse each B fragment.
    for (int ntI = 0; ntI < 4; ++ntI) {
        int ncol = (wave + 8 * ntI) * 16;     // 32 N-tiles over 8 waves x 4
        v8f ag[6], ah[6];
        v8f zero = {};
        #pragma unroll
        for (int m = 0; m < 6; ++m) { ag[m] = zero; ah[m] = zero; }
        for (int kb = 0; kb < CINW; kb += 32) {
            v16bf b1 = load_b_frag(Wgt, CINW, ncol, kb, l, hi);
            v16bf b2 = load_b_frag(Wht, CINW, ncol, kb, l, hi);
            #pragma unroll
            for (int m = 0; m < 6; ++m) {
                v16bf a = load_a_frag(feats, FS, m * 16, kb, l, hi);
                ag[m] = wmma_bf16(a, b1, ag[m]);
                ah[m] = wmma_bf16(a, b2, ah[m]);
            }
        }
        float bgv = bg[ncol + l], bhv = bh[ncol + l];
        #pragma unroll
        for (int m = 0; m < 6; ++m) {
            #pragma unroll
            for (int p = 0; p < 8; ++p) {
                float gv = ag[m][p] + bgv, hv = ah[m][p] + bhv;
                act[(m * 16 + p + hi * 8) * AS + ncol + l] = (bf16_t)(gelu_tanh(gv) * hv);
            }
        }
    }
    __syncthreads();

    // ---- GEMM 2: upd = act @ Wo + bo (f32, overlays feats)
    for (int ntI = 0; ntI < 2; ++ntI) {
        int ncol = (wave + 8 * ntI) * 16;     // 16 N-tiles over 8 waves x 2
        v8f acc[6];
        v8f zero = {};
        #pragma unroll
        for (int m = 0; m < 6; ++m) acc[m] = zero;
        for (int kb = 0; kb < HH; kb += 32) {
            v16bf b = load_b_frag(Wot, HH, ncol, kb, l, hi);
            #pragma unroll
            for (int m = 0; m < 6; ++m) {
                v16bf a = load_a_frag(act, AS, m * 16, kb, l, hi);
                acc[m] = wmma_bf16(a, b, acc[m]);
            }
        }
        float bov = bo[ncol + l];
        #pragma unroll
        for (int m = 0; m < 6; ++m) {
            #pragma unroll
            for (int p = 0; p < 8; ++p)
                upd[(m * 16 + p + hi * 8) * US + ncol + l] = acc[m][p] + bov;
        }
    }
    __syncthreads();

    if (phase == 0) {
        // residual + LN per row -> pair_new / extra_new
        for (int rr2 = 0; rr2 < 12; ++rr2) {
            int row = wave * 12 + rr2;
            int g = (row >= ROWS) ? 1 : 0;
            int r = row - g * ROWS;
            int i = i0 + g;
            const float* src = (r < KNB)
                ? pair_in + ((size_t)i * KNB + r) * DD
                : extra_in + ((size_t)i * ENB + (r - KNB)) * DD;
            float xs[8], s = 0.f, ss = 0.f;
            for (int q = 0; q < 8; ++q) {
                int c = lane + q * 32;
                float x = src[c] + upd[row * US + c];
                xs[q] = x; s += x; ss += x * x;
            }
            s = wave_sum(s); ss = wave_sum(ss);
            float mean = s * (1.0f / 256.0f);
            float var  = ss * (1.0f / 256.0f) - mean * mean;
            float inv  = rsqrtf(var + 1e-5f);
            float* dst = (r < KNB)
                ? out_pair + ((size_t)i * KNB + r) * DD
                : out_extra + ((size_t)i * ENB + (r - KNB)) * DD;
            for (int q = 0; q < 8; ++q) {
                int c = lane + q * 32;
                dst[c] = (xs[q] - mean) * inv * ln_s[c] + ln_b[c];
            }
        }
    } else {
        // masked mean over 48 rows per node + residual + LN -> local1
        if (tid < ROWS2) {
            int g = (tid >= ROWS) ? 1 : 0;
            int r = tid - g * ROWS;
            int i = i0 + g;
            float ok;
            if (r < KNB) {
                int nbv = neighbours[(size_t)i * KNB + r];
                ok = (nbv != -1 && mask[nbv] != 0) ? 1.0f : 0.0f;
            } else {
                ok = (extra_mask[(size_t)i * ENB + (r - KNB)] != 0) ? 1.0f : 0.0f;
            }
            rmask[tid] = ok;
        }
        __syncthreads();
        for (int g = 0; g < GNODE; ++g) {
            int c = tid;
            float msg = 0.f;
            for (int r = 0; r < ROWS; ++r)
                msg += rmask[g * ROWS + r] * upd[(g * ROWS + r) * US + c];
            msg *= (1.0f / (float)ROWS);
            float x = local_in[(size_t)(i0 + g) * DD + c] + msg;
            float s = wave_sum(x), ss = wave_sum(x * x);
            if (lane == 0) { red[wave] = s; red[8 + wave] = ss; }
            __syncthreads();
            float S = 0.f, SS = 0.f;
            for (int w = 0; w < 8; ++w) { S += red[w]; SS += red[8 + w]; }
            float mean = S * (1.0f / 256.0f);
            float var  = SS * (1.0f / 256.0f) - mean * mean;
            float inv  = rsqrtf(var + 1e-5f);
            out_local1[(size_t)(i0 + g) * DD + c] = (x - mean) * inv * ln_s[c] + ln_b[c];
            __syncthreads();   // red reused next g
        }
    }
}

// ---------------------------------------------------------------------------
// g3 gated MLP over [N,256]; 32 rows per workgroup, B reused across 2 M-tiles.
// ---------------------------------------------------------------------------
__global__ __launch_bounds__(256) void local_mlp_kernel(
    const float* __restrict__ x_in,    // local1 [N,256]
    const bf16_t* __restrict__ Wgt,    // [512][256]
    const bf16_t* __restrict__ Wht,    // [512][256]
    const bf16_t* __restrict__ Wot,    // [256][512]
    const float* __restrict__ bg, const float* __restrict__ bh,
    const float* __restrict__ bo,
    const float* __restrict__ ln_s, const float* __restrict__ ln_b,
    float* __restrict__ out)
{
    extern __shared__ __align__(16) char smem[];
    bf16_t* xb  = (bf16_t*)smem;                 // 32 x XS bf16 (17408 B)
    bf16_t* act = (bf16_t*)(smem + 17408);       // 32 x AS bf16 (33792 B)
    float*  upd = (float*)(smem + 51200);        // 32 x US f32  (33280 B)

    const int r0 = blockIdx.x * 32;
    const int tid = threadIdx.x, wave = tid >> 5, lane = tid & 31;
    const int l = lane & 15, hi = lane >> 4;

    for (int e = tid; e < 32 * DD; e += 256) {
        int r = e >> 8, c = e & 255;
        xb[r * XS + c] = (bf16_t)x_in[(size_t)(r0 + r) * DD + c];
    }
    __syncthreads();

    for (int ntI = 0; ntI < 4; ++ntI) {
        int ncol = (wave + 8 * ntI) * 16;        // 32 N-tiles
        v8f ag[2], ah[2];
        v8f zero = {};
        #pragma unroll
        for (int m = 0; m < 2; ++m) { ag[m] = zero; ah[m] = zero; }
        for (int kb = 0; kb < DD; kb += 32) {
            v16bf b1 = load_b_frag(Wgt, DD, ncol, kb, l, hi);
            v16bf b2 = load_b_frag(Wht, DD, ncol, kb, l, hi);
            #pragma unroll
            for (int m = 0; m < 2; ++m) {
                v16bf a = load_a_frag(xb, XS, m * 16, kb, l, hi);
                ag[m] = wmma_bf16(a, b1, ag[m]);
                ah[m] = wmma_bf16(a, b2, ah[m]);
            }
        }
        float bgv = bg[ncol + l], bhv = bh[ncol + l];
        #pragma unroll
        for (int m = 0; m < 2; ++m) {
            #pragma unroll
            for (int p = 0; p < 8; ++p) {
                float gv = ag[m][p] + bgv, hv = ah[m][p] + bhv;
                act[(m * 16 + p + hi * 8) * AS + ncol + l] = (bf16_t)(gelu_tanh(gv) * hv);
            }
        }
    }
    __syncthreads();

    for (int ntI = 0; ntI < 2; ++ntI) {
        int ncol = (wave + 8 * ntI) * 16;        // 16 N-tiles
        v8f acc[2];
        v8f zero = {};
        #pragma unroll
        for (int m = 0; m < 2; ++m) acc[m] = zero;
        for (int kb = 0; kb < HH; kb += 32) {
            v16bf b = load_b_frag(Wot, HH, ncol, kb, l, hi);
            #pragma unroll
            for (int m = 0; m < 2; ++m) {
                v16bf a = load_a_frag(act, AS, m * 16, kb, l, hi);
                acc[m] = wmma_bf16(a, b, acc[m]);
            }
        }
        float bov = bo[ncol + l];
        #pragma unroll
        for (int m = 0; m < 2; ++m) {
            #pragma unroll
            for (int p = 0; p < 8; ++p)
                upd[(m * 16 + p + hi * 8) * US + ncol + l] = acc[m][p] + bov;
        }
    }
    __syncthreads();

    for (int rr = 0; rr < 4; ++rr) {
        int row = wave * 4 + rr;
        const float* src = x_in + (size_t)(r0 + row) * DD;
        float xs[8], s = 0.f, ss = 0.f;
        for (int q = 0; q < 8; ++q) {
            int c = lane + q * 32;
            float x = src[c] + upd[row * US + c];
            xs[q] = x; s += x; ss += x * x;
        }
        s = wave_sum(s); ss = wave_sum(ss);
        float mean = s * (1.0f / 256.0f);
        float var  = ss * (1.0f / 256.0f) - mean * mean;
        float inv  = rsqrtf(var + 1e-5f);
        float* dst = out + (size_t)(r0 + row) * DD;
        for (int q = 0; q < 8; ++q) {
            int c = lane + q * 32;
            dst[c] = (xs[q] - mean) * inv * ln_s[c] + ln_b[c];
        }
    }
}

// ---------------------------------------------------------------------------
extern "C" void kernel_launch(void* const* d_in, const int* in_sizes, int n_in,
                              void* d_out, int out_size, void* d_ws, size_t ws_size,
                              hipStream_t stream) {
    (void)in_sizes; (void)n_in; (void)out_size; (void)ws_size;

    const float* local      = (const float*)d_in[0];
    const float* pair       = (const float*)d_in[1];
    const float* extra_pair = (const float*)d_in[4];
    const unsigned char* extra_mask = (const unsigned char*)d_in[5];
    const int* neighbours   = (const int*)d_in[6];
    const unsigned char* mask = (const unsigned char*)d_in[7];
    const float* W_ep1 = (const float*)d_in[8];
    const float* W_ep2 = (const float*)d_in[9];
    const float* g1_Wg = (const float*)d_in[10];
    const float* g1_bg = (const float*)d_in[11];
    const float* g1_Wh = (const float*)d_in[12];
    const float* g1_bh = (const float*)d_in[13];
    const float* g1_Wo = (const float*)d_in[14];
    const float* g1_bo = (const float*)d_in[15];
    const float* g2_Wg = (const float*)d_in[16];
    const float* g2_bg = (const float*)d_in[17];
    const float* g2_Wh = (const float*)d_in[18];
    const float* g2_bh = (const float*)d_in[19];
    const float* g2_Wo = (const float*)d_in[20];
    const float* g2_bo = (const float*)d_in[21];
    const float* g3_Wg = (const float*)d_in[22];
    const float* g3_bg = (const float*)d_in[23];
    const float* g3_Wh = (const float*)d_in[24];
    const float* g3_bh = (const float*)d_in[25];
    const float* g3_Wo = (const float*)d_in[26];
    const float* g3_bo = (const float*)d_in[27];
    const float* ln_fp_s = (const float*)d_in[28];
    const float* ln_fp_b = (const float*)d_in[29];
    const float* ln_l1_s = (const float*)d_in[30];
    const float* ln_l1_b = (const float*)d_in[31];
    const float* ln_l2_s = (const float*)d_in[32];
    const float* ln_l2_b = (const float*)d_in[33];

    float* out_local = (float*)d_out;
    float* out_pair  = out_local + (size_t)NNODES * DD;
    float* out_extra = out_pair + (size_t)NNODES * KNB * DD;

    // workspace carve
    char* w = (char*)d_ws;
    auto take = [&](size_t bytes) { char* p = w; w += bytes; return p; };
    bf16_t* Wg1t  = (bf16_t*)take((size_t)HH * CINW * 2);
    bf16_t* Wh1t  = (bf16_t*)take((size_t)HH * CINW * 2);
    bf16_t* Wo1t  = (bf16_t*)take((size_t)DD * HH * 2);
    bf16_t* Wg2t  = (bf16_t*)take((size_t)HH * CINW * 2);
    bf16_t* Wh2t  = (bf16_t*)take((size_t)HH * CINW * 2);
    bf16_t* Wo2t  = (bf16_t*)take((size_t)DD * HH * 2);
    bf16_t* Wg3t  = (bf16_t*)take((size_t)HH * DD * 2);
    bf16_t* Wh3t  = (bf16_t*)take((size_t)HH * DD * 2);
    bf16_t* Wo3t  = (bf16_t*)take((size_t)DD * HH * 2);
    bf16_t* Wep1t = (bf16_t*)take((size_t)DD * DD * 2);
    bf16_t* Wep2t = (bf16_t*)take((size_t)DD * DD * 2);
    float*  local1 = (float*)take((size_t)NNODES * DD * 4);

    auto conv = [&](const float* s, bf16_t* d, int in_dim, int out_dim) {
        int total = in_dim * out_dim;
        convert_transpose_kernel<<<(total + 255) / 256, 256, 0, stream>>>(
            s, d, in_dim, out_dim);
    };
    conv(g1_Wg, Wg1t, CINW, HH);
    conv(g1_Wh, Wh1t, CINW, HH);
    conv(g1_Wo, Wo1t, HH, DD);
    conv(g2_Wg, Wg2t, CINW, HH);
    conv(g2_Wh, Wh2t, CINW, HH);
    conv(g2_Wo, Wo2t, HH, DD);
    conv(g3_Wg, Wg3t, DD, HH);
    conv(g3_Wh, Wh3t, DD, HH);
    conv(g3_Wo, Wo3t, HH, DD);
    conv(W_ep1, Wep1t, DD, DD);
    conv(W_ep2, Wep2t, DD, DD);

    mpnn_pass_kernel<<<NNODES / GNODE, 256, SMEM_PASS, stream>>>(
        local, pair, extra_pair, neighbours, mask, extra_mask,
        Wep1t, Wg1t, Wh1t, Wo1t, g1_bg, g1_bh, g1_bo,
        ln_fp_s, ln_fp_b, out_pair, out_extra, nullptr, 0);

    mpnn_pass_kernel<<<NNODES / GNODE, 256, SMEM_PASS, stream>>>(
        local, out_pair, out_extra, neighbours, mask, extra_mask,
        Wep2t, Wg2t, Wh2t, Wo2t, g2_bg, g2_bh, g2_bo,
        ln_l1_s, ln_l1_b, nullptr, nullptr, local1, 1);

    local_mlp_kernel<<<NNODES / 32, 256, SMEM_MLP, stream>>>(
        local1, Wg3t, Wh3t, Wo3t, g3_bg, g3_bh, g3_bo,
        ln_l2_s, ln_l2_b, out_local);
}
